// TenNet_Tag_57105885167624
// MI455X (gfx1250) — matrix operand order repaired and checked by reading, not compiled, not run on hardware
//
#include <hip/hip_runtime.h>
#include <hip/hip_bf16.h>

// CDNA5 / gfx1250 implementation of TenNet_Tag forward pass.
// All convolutions + FC layers run on v_wmma_f32_16x16x32_f16 (wave32 WMMA).
// Activations kept in f16 (bandwidth-bound). Conv inputs are staged per
// workgroup into LDS (<=12.3KB) so the im2col A-gather is ds_load based and
// the k^2-redundant reads never hit the global path.

typedef _Float16 h16;
typedef __attribute__((ext_vector_type(16))) _Float16 v16h;
typedef __attribute__((ext_vector_type(8)))  float    v8f;

// ---------------- weight packing: f32 OIHW -> f16 [CoutPad][Kpad] ----------
__global__ __launch_bounds__(256) void pack_w_kernel(
    const float* __restrict__ w, const float* __restrict__ bias,
    h16* __restrict__ wpk, float* __restrict__ bpk,
    int cout, int K, int coutPad, int Kpad)
{
  int idx = blockIdx.x * blockDim.x + threadIdx.x;
  int total = coutPad * Kpad;
  if (idx < total) {
    int co = idx / Kpad;
    int k  = idx - co * Kpad;
    float v = (co < cout && k < K) ? w[co * K + k] : 0.0f;
    wpk[idx] = (h16)v;
  }
  if (idx < coutPad)
    bpk[idx] = (idx < cout) ? bias[idx] : 0.0f;
}

// ---------------- embedding scatter -> f16 image [8][2][V][D] --------------
__global__ __launch_bounds__(128) void embed_kernel(
    const int* __restrict__ tags, const float* __restrict__ Wu,
    h16* __restrict__ X0, int V, int D)
{
  int v = blockIdx.x, b = blockIdx.y;
  bool on = tags[b * V + v] > 0;
  size_t plane = (size_t)V * D;
  size_t base0 = ((size_t)b * 2) * plane + (size_t)v * D;
  for (int d = threadIdx.x; d < D; d += blockDim.x) {
    h16 val = on ? (h16)Wu[(size_t)v * D + d] : (h16)0.0f;
    X0[base0 + d]         = val;   // channel 0
    X0[base0 + plane + d] = val;   // channel 1 (reference duplicates)
  }
}

// ---------------- 3x3 maxpool, pad 1, stride 1 or 2 ------------------------
__global__ __launch_bounds__(256) void maxpool_kernel(
    const h16* __restrict__ X, h16* __restrict__ Y,
    int C, int H, int W, int Ho, int Wo, int stride, int total)
{
  int idx = blockIdx.x * blockDim.x + threadIdx.x;
  if (idx >= total) return;
  int wo = idx % Wo; int t = idx / Wo;
  int ho = t % Ho;   t /= Ho;
  int c  = t % C;    int b = t / C;
  const h16* src = X + ((size_t)b * C + c) * H * W;
  float mx = -1e30f;
  #pragma unroll
  for (int dy = 0; dy < 3; ++dy) {
    int iy = ho * stride + dy - 1;
    if (iy < 0 || iy >= H) continue;
    #pragma unroll
    for (int dx = 0; dx < 3; ++dx) {
      int ix = wo * stride + dx - 1;
      if (ix < 0 || ix >= W) continue;
      mx = fmaxf(mx, (float)src[(size_t)iy * W + ix]);
    }
  }
  Y[idx] = (h16)mx;
}

// ---------------- implicit-GEMM conv via v_wmma_f32_16x16x32_f16 -----------
// Workgroup = 8 waves = 128 consecutive output pixels for one (batch, co16).
// Input slab for those pixels is staged into LDS once (coalesced), then each
// wave builds its 16x32 A fragment from LDS and runs WMMA against contiguous
// 32B packed-weight B fragments.  Epilogue fuses bias + ReLU + concat offset.
template <int KSZ>
__global__ __launch_bounds__(256) void conv_wmma_kernel(
    const h16* __restrict__ X,    // [8][Cin][H][W]
    const h16* __restrict__ Wpk,  // [CoutPad][Kpad] f16
    const float* __restrict__ Bpk,// [CoutPad]
    h16* __restrict__ Y,          // [8][CoutTot][H][W]
    int Cin, int H, int W, int Kpad,
    int cout, int CoutTot, int coff, int relu)
{
  constexpr int PAD = KSZ / 2;
  constexpr int KK  = KSZ * KSZ;
  __shared__ h16 slab[6144];      // 12.3KB: max(Cin*128, Cin*R*W) by config

  const int HW = H * W;
  const int b  = blockIdx.z;
  const h16* Xb = X + (size_t)b * Cin * HW;
  const int p0 = blockIdx.x * 128;

  int R = 0, row0 = 0;
  if (KSZ == 1) {
    // slab[ci][x] = X[ci][p0+x], zero-padded past HW
    const int tot = Cin << 7;
    for (int idx = threadIdx.x; idx < tot; idx += 256) {
      int ci = idx >> 7;
      int x  = idx & 127;
      int m  = p0 + x;
      slab[idx] = (m < HW) ? Xb[(size_t)ci * HW + m] : (h16)0.0f;
    }
  } else {
    // rows [row0, rowLast] cover every (pixel +/- PAD) this block touches;
    // vertical zero-padding is materialized here so the gather skips iy tests
    int pLast = p0 + 127; if (pLast >= HW) pLast = HW - 1;
    row0 = p0 / W - PAD;
    int rowLast = pLast / W + PAD;
    R = rowLast - row0 + 1;
    const int tot = Cin * R * W;
    for (int idx = threadIdx.x; idx < tot; idx += 256) {
      int ciR = idx / W;
      int x   = idx - ciR * W;
      int ci  = ciR / R;
      int r   = ciR - ci * R;
      int iy  = row0 + r;
      slab[idx] = (iy >= 0 && iy < H) ? Xb[((size_t)ci * H + iy) * W + x]
                                      : (h16)0.0f;
    }
  }
  __syncthreads();

  const int pixTiles = (HW + 15) >> 4;
  const int wid  = threadIdx.x >> 5;
  const int tile = blockIdx.x * 8 + wid;
  if (tile >= pixTiles) return;                 // wave-uniform

  const int l  = threadIdx.x & 31;
  const int g  = l >> 4;
  const int nl = l & 15;            // A-row (M) this lane feeds; N for B/C/D
  const int coBase = blockIdx.y * 16;
  const int co = coBase + nl;
  const int m0 = tile * 16;
  const int Kreal = Cin * KK;

  v8f acc;
  {
    float bv = Bpk[co];
    #pragma unroll
    for (int r = 0; r < 8; ++r) acc[r] = bv;
  }

  const h16* wrow = Wpk + (size_t)co * Kpad;
  __builtin_prefetch(wrow, 0, 3);               // global_prefetch_b8

  const int  m  = m0 + nl;
  const bool mv = (m < HW);
  int oy = 0, ox = 0, locP = 0;
  if (KSZ == 1) {
    locP = (wid << 4) + nl;                     // m - p0
  } else {
    oy = mv ? (m / W) : 0;
    ox = m - oy * W;
  }

  for (int k0 = 0; k0 < Kpad; k0 += 32) {
    // B fragment: lane holds B[K=16g+e][N=nl] -> 16 contiguous packed halves
    v16h bf = *(const v16h*)(wrow + k0 + 16 * g);
    // A fragment from LDS slab: lane holds A[M=nl][K=16g+e], matching B
    v16h af;
    #pragma unroll
    for (int e = 0; e < 16; ++e) {
      int k = k0 + 16 * g + e;
      h16 v = (h16)0.0f;
      if (mv && k < Kreal) {
        if (KSZ == 1) {
          v = slab[(k << 7) + locP];
        } else {
          int ci  = k / KK;
          int rem = k - ci * KK;
          int ky  = rem / KSZ;
          int kx  = rem - ky * KSZ;
          int ix  = ox + kx - PAD;
          if (ix >= 0 && ix < W) {
            int iy = oy + ky - PAD;             // slab row always in range
            v = slab[(ci * R + (iy - row0)) * W + ix];
          }
        }
      }
      af[e] = v;
    }
    acc = __builtin_amdgcn_wmma_f32_16x16x32_f16(false, af, false, bf,
                                                 (short)0, acc, false, false);
  }

  // D layout: lane l holds N=nl, VGPR r holds pixel m0 + r + 8g
  if (co < cout) {
    h16* ydst = Y + ((size_t)b * CoutTot + (coff + co)) * HW;
    int mbase = m0 + 8 * g;
    if (m0 + 16 <= HW) {
      unsigned pk[4];
      #pragma unroll
      for (int r = 0; r < 8; r += 2) {
        float v0 = acc[r], v1 = acc[r + 1];
        if (relu) { v0 = fmaxf(v0, 0.0f); v1 = fmaxf(v1, 0.0f); }
        unsigned u0 = (unsigned)__builtin_bit_cast(unsigned short, (h16)v0);
        unsigned u1 = (unsigned)__builtin_bit_cast(unsigned short, (h16)v1);
        pk[r >> 1] = u0 | (u1 << 16);
      }
      unsigned* yd32 = (unsigned*)(ydst + mbase);
      #pragma unroll
      for (int q = 0; q < 4; ++q) yd32[q] = pk[q];
    } else {
      #pragma unroll
      for (int r = 0; r < 8; ++r) {
        int mm = mbase + r;
        if (mm < HW) {
          float v = acc[r];
          if (relu) v = fmaxf(v, 0.0f);
          ydst[mm] = (h16)v;
        }
      }
    }
  }
}

// ---------------- FC layer (M=16 incl. 8 pad rows) via WMMA ----------------
__global__ __launch_bounds__(32) void fc_wmma_kernel(
    const float* __restrict__ actIn,  // [16][Kpad] f32 (zero-padded)
    const h16* __restrict__ Wpk,      // [CoutPad][Kpad] f16
    const float* __restrict__ Bpk,
    const float* __restrict__ bnB, const float* __restrict__ bnG,
    const float* __restrict__ bnM, const float* __restrict__ bnV,
    float* __restrict__ actOut,       // [16][NoutStride]
    int Kpad, int cout, int NoutStride, int relu)
{
  const int l  = threadIdx.x & 31;
  const int g  = l >> 4;
  const int nl = l & 15;
  const int co = blockIdx.x * 16 + nl;

  v8f acc;
  {
    float bv = Bpk[co];
    #pragma unroll
    for (int r = 0; r < 8; ++r) acc[r] = bv;
  }

  const float* arow = actIn + (size_t)nl * Kpad;
  const h16*   wrow = Wpk + (size_t)co * Kpad;
  __builtin_prefetch(wrow, 0, 3);

  for (int k0 = 0; k0 < Kpad; k0 += 32) {
    v16h bf = *(const v16h*)(wrow + k0 + 16 * g);
    v16h af;
    #pragma unroll
    for (int e = 0; e < 16; ++e) {
      int k = k0 + 16 * g + e;
      af[e] = (h16)arow[k];
    }
    acc = __builtin_amdgcn_wmma_f32_16x16x32_f16(false, af, false, bf,
                                                 (short)0, acc, false, false);
  }

  bool cv = (co < cout);
  float scale = 1.0f, shift = 0.0f;
  if (cv && bnG) {
    float inv = rsqrtf(bnV[co] + 1e-5f);
    scale = bnG[co] * inv;
    shift = bnB[co] - bnM[co] * scale;
  }
  #pragma unroll
  for (int r = 0; r < 8; ++r) {
    int row = r + 8 * g;
    float v = acc[r];
    if (bnG)  v = v * scale + shift;
    if (relu) v = fmaxf(v, 0.0f);
    actOut[(size_t)row * NoutStride + co] = cv ? v : 0.0f;
  }
}

// ---------------- global average pool: [8][128][HW] -> act[16][128] --------
__global__ __launch_bounds__(256) void gap_kernel(
    const h16* __restrict__ Y, float* __restrict__ act0,
    int C, int HW, int Kstride)
{
  int c = blockIdx.x, b = blockIdx.y;
  const h16* src = Y + ((size_t)b * C + c) * HW;
  float s = 0.0f;
  for (int i = threadIdx.x; i < HW; i += blockDim.x) s += (float)src[i];
  __shared__ float red[256];
  red[threadIdx.x] = s;
  __syncthreads();
  for (int st = 128; st > 0; st >>= 1) {
    if (threadIdx.x < st) red[threadIdx.x] += red[threadIdx.x + st];
    __syncthreads();
  }
  if (threadIdx.x == 0) act0[(size_t)b * Kstride + c] = red[0] / (float)HW;
}

__global__ __launch_bounds__(256) void zero_f32_kernel(float* p, int n) {
  int i = blockIdx.x * blockDim.x + threadIdx.x;
  if (i < n) p[i] = 0.0f;
}

// ---------------- global min-max normalize over [8][128] -------------------
__global__ __launch_bounds__(256) void minmax_kernel(
    const float* __restrict__ act3, float* __restrict__ out, int n)
{
  __shared__ float smin[256], smax[256];
  float lo = 1e30f, hi = -1e30f;
  for (int i = threadIdx.x; i < n; i += 256) {
    float v = act3[i];
    lo = fminf(lo, v); hi = fmaxf(hi, v);
  }
  smin[threadIdx.x] = lo; smax[threadIdx.x] = hi;
  __syncthreads();
  for (int st = 128; st > 0; st >>= 1) {
    if (threadIdx.x < st) {
      smin[threadIdx.x] = fminf(smin[threadIdx.x], smin[threadIdx.x + st]);
      smax[threadIdx.x] = fmaxf(smax[threadIdx.x], smax[threadIdx.x + st]);
    }
    __syncthreads();
  }
  float l0 = smin[0], h0 = smax[0];
  float inv = 1.0f / (h0 - l0);
  for (int i = threadIdx.x; i < n; i += 256)
    out[i] = (act3[i] - l0) * inv;
}

// ===========================================================================
extern "C" void kernel_launch(void* const* d_in, const int* in_sizes, int n_in,
                              void* d_out, int out_size, void* d_ws, size_t ws_size,
                              hipStream_t stream)
{
  (void)in_sizes; (void)n_in; (void)out_size; (void)ws_size;
  const int B = 8, V = 2000, D = 300;

  // JAX pytree flatten order (sorted dict keys): params{Wu, fc{W1,W2,W3,b1,
  // b2,b3,bn1{b,g,m,v},bn2{b,g,m,v}}, inc[i]{b1,b2a,b2b,b3a,b3b,b4,w1,w2a,
  // w2b,w3a,w3b,w4}}, then tags.
  const float* Wu     = (const float*)d_in[0];
  const float* fcW[3] = {(const float*)d_in[1], (const float*)d_in[2], (const float*)d_in[3]};
  const float* fcb[3] = {(const float*)d_in[4], (const float*)d_in[5], (const float*)d_in[6]};
  const float* bn1[4] = {(const float*)d_in[7],  (const float*)d_in[8],
                         (const float*)d_in[9],  (const float*)d_in[10]}; // b,g,m,v
  const float* bn2[4] = {(const float*)d_in[11], (const float*)d_in[12],
                         (const float*)d_in[13], (const float*)d_in[14]};
  const int* tags = (const int*)d_in[63];

  char* ws = (char*)d_ws;
  size_t off = 0;
  auto balloc = [&](size_t bytes) { size_t o = off; off += (bytes + 255) & ~255ULL; return o; };

  size_t offX = balloc(48000000);   // block input ping   (f16)
  size_t offY = balloc(192000000);  // block output       (f16)
  size_t offT = balloc(48000000);   // branch temporaries (f16)

  size_t offF = balloc((size_t)16 * (128 + 512 + 1056 + 128) * sizeof(float));
  float* act0 = (float*)(ws + offF);
  float* act1 = act0 + 16 * 128;
  float* act2 = act1 + 16 * 512;
  float* act3 = act2 + 16 * 1056;

  const int cfg[4][7] = {  // cin, c1, c2a, c2b, c3a, c3b, c4
    { 2,  4, 4,  8, 2,  4,  4},
    {20,  8, 8, 16, 2,  4,  4},
    {32,  8, 8, 16, 8, 16,  8},
    {48, 32, 8, 32, 8, 32, 32}};
  const int Hs[4]  = {2000, 1000, 500, 250};
  const int Wsp[4] = {300, 150, 75, 38};

  auto ceilN = [](int x, int n) { return (x + n - 1) / n * n; };

  struct PW { const float* w; const float* b; int cout, K, Kpad, CoutPad; size_t wOff, bOff; };
  PW pw[27]; int npw = 0;
  auto addPW = [&](const float* w, const float* b, int cout, int K) {
    PW p; p.w = w; p.b = b; p.cout = cout; p.K = K;
    p.Kpad = ceilN(K, 32); p.CoutPad = ceilN(cout, 16);
    p.wOff = balloc((size_t)p.CoutPad * p.Kpad * sizeof(h16));
    p.bOff = balloc((size_t)p.CoutPad * sizeof(float));
    pw[npw] = p;
    return npw++;
  };

  int cvW[4][6];
  for (int i = 0; i < 4; ++i) {
    int LB = 15 + 12 * i;
    int cin = cfg[i][0], c2a = cfg[i][2], c3a = cfg[i][4];
    cvW[i][0] = addPW((const float*)d_in[LB + 6],  (const float*)d_in[LB + 0], cfg[i][1], cin);
    cvW[i][1] = addPW((const float*)d_in[LB + 7],  (const float*)d_in[LB + 1], c2a, cin);
    cvW[i][2] = addPW((const float*)d_in[LB + 8],  (const float*)d_in[LB + 2], cfg[i][3], c2a * 9);
    cvW[i][3] = addPW((const float*)d_in[LB + 9],  (const float*)d_in[LB + 3], c3a, cin);
    cvW[i][4] = addPW((const float*)d_in[LB + 10], (const float*)d_in[LB + 4], cfg[i][5], c3a * 25);
    cvW[i][5] = addPW((const float*)d_in[LB + 11], (const float*)d_in[LB + 5], cfg[i][6], cin);
  }
  int fcI[3];
  fcI[0] = addPW(fcW[0], fcb[0], 512, 128);
  fcI[1] = addPW(fcW[1], fcb[1], 1028, 512);
  fcI[2] = addPW(fcW[2], fcb[2], 128, 1028);

  for (int i = 0; i < npw; ++i) {
    int total = pw[i].CoutPad * pw[i].Kpad;
    pack_w_kernel<<<(total + 255) / 256, 256, 0, stream>>>(
        pw[i].w, pw[i].b, (h16*)(ws + pw[i].wOff), (float*)(ws + pw[i].bOff),
        pw[i].cout, pw[i].K, pw[i].CoutPad, pw[i].Kpad);
  }

  {
    int n = 16 * (128 + 512 + 1056 + 128);
    zero_f32_kernel<<<(n + 255) / 256, 256, 0, stream>>>(act0, n);
  }

  h16* Xbuf = (h16*)(ws + offX);
  h16* Ybuf = (h16*)(ws + offY);
  h16* Tbuf = (h16*)(ws + offT);

  embed_kernel<<<dim3(V, B), 128, 0, stream>>>(tags, Wu, Xbuf, V, D);

  auto launchConv = [&](const h16* X, int pwi, h16* Y, int Cin, int H, int W,
                        int k, int CoutTot, int coff) {
    const PW& p = pw[pwi];
    int tiles = (H * W + 15) / 16;
    dim3 grid((tiles + 7) / 8, p.CoutPad / 16, B);
    const h16* wp = (const h16*)(ws + p.wOff);
    const float* bp = (const float*)(ws + p.bOff);
    if (k == 1)
      conv_wmma_kernel<1><<<grid, 256, 0, stream>>>(X, wp, bp, Y, Cin, H, W,
          p.Kpad, p.cout, CoutTot, coff, 1);
    else if (k == 3)
      conv_wmma_kernel<3><<<grid, 256, 0, stream>>>(X, wp, bp, Y, Cin, H, W,
          p.Kpad, p.cout, CoutTot, coff, 1);
    else
      conv_wmma_kernel<5><<<grid, 256, 0, stream>>>(X, wp, bp, Y, Cin, H, W,
          p.Kpad, p.cout, CoutTot, coff, 1);
  };

  for (int i = 0; i < 4; ++i) {
    int cin = cfg[i][0], c1 = cfg[i][1], c2a = cfg[i][2], c2b = cfg[i][3],
        c3a = cfg[i][4], c3b = cfg[i][5], c4 = cfg[i][6];
    int H = Hs[i], W = Wsp[i];
    int CoutTot = c1 + c2b + c3b + c4;
    (void)c4;

    // branch 1: 1x1
    launchConv(Xbuf, cvW[i][0], Ybuf, cin, H, W, 1, CoutTot, 0);
    // branch 2: 1x1 -> 3x3
    launchConv(Xbuf, cvW[i][1], Tbuf, cin, H, W, 1, c2a, 0);
    launchConv(Tbuf, cvW[i][2], Ybuf, c2a, H, W, 3, CoutTot, c1);
    // branch 3: 1x1 -> 5x5
    launchConv(Xbuf, cvW[i][3], Tbuf, cin, H, W, 1, c3a, 0);
    launchConv(Tbuf, cvW[i][4], Ybuf, c3a, H, W, 5, CoutTot, c1 + c2b);
    // branch 4: maxpool(3, s1) -> 1x1
    {
      int total = B * cin * H * W;
      maxpool_kernel<<<(total + 255) / 256, 256, 0, stream>>>(
          Xbuf, Tbuf, cin, H, W, H, W, 1, total);
      launchConv(Tbuf, cvW[i][5], Ybuf, cin, H, W, 1, CoutTot, c1 + c2b + c3b);
    }
    // downsample between blocks
    if (i < 3) {
      int Ho = Hs[i + 1], Wo = Wsp[i + 1];
      int total = B * CoutTot * Ho * Wo;
      maxpool_kernel<<<(total + 255) / 256, 256, 0, stream>>>(
          Ybuf, Xbuf, CoutTot, H, W, Ho, Wo, 2, total);
    }
  }

  // Global average pool: Ybuf [8][128][250][38] -> act0 rows 0..7
  gap_kernel<<<dim3(128, B), 256, 0, stream>>>(Ybuf, act0, 128, 250 * 38, 128);

  // FC stack with fused BN + ReLU
  {
    const PW& p1 = pw[fcI[0]];
    fc_wmma_kernel<<<p1.CoutPad / 16, 32, 0, stream>>>(act0,
        (const h16*)(ws + p1.wOff), (const float*)(ws + p1.bOff),
        bn1[0], bn1[1], bn1[2], bn1[3], act1, p1.Kpad, 512, 512, 1);
    const PW& p2 = pw[fcI[1]];
    fc_wmma_kernel<<<p2.CoutPad / 16, 32, 0, stream>>>(act1,
        (const h16*)(ws + p2.wOff), (const float*)(ws + p2.bOff),
        bn2[0], bn2[1], bn2[2], bn2[3], act2, p2.Kpad, 1028, 1056, 1);
    const PW& p3 = pw[fcI[2]];
    fc_wmma_kernel<<<p3.CoutPad / 16, 32, 0, stream>>>(act2,
        (const h16*)(ws + p3.wOff), (const float*)(ws + p3.bOff),
        nullptr, nullptr, nullptr, nullptr, act3, p3.Kpad, 128, 128, 0);
  }

  minmax_kernel<<<1, 256, 0, stream>>>(act3, (float*)d_out, 8 * 128);
}
